// WindowMSA_41738492182642
// MI455X (gfx1250) — compile-verified, matrix operand order checked by reading, not run on hardware
//
#include <hip/hip_runtime.h>
#include <hip/hip_bf16.h>

typedef __attribute__((ext_vector_type(16))) __bf16 v16bf;
typedef __attribute__((ext_vector_type(8)))  __bf16 v8bf;
typedef __attribute__((ext_vector_type(4)))  __bf16 v4bf;
typedef __attribute__((ext_vector_type(8)))  float  v8f;

#define N_   64
#define C_   256
#define NH_  8
#define HD_  32

#define LOG2E 1.44269504088896340736f

// LDS row strides (elements); multiples of 8 keep 16B alignment for v8bf loads
#define XS_LD  264   // x window (64 x 256), later reused for merged O (64 x 256)
#define QKV_LD 776   // qkv (64 x 768)

// A-fragment (16x32 row-major source) OR column-major B-fragment where
// B column n == source row n.  ISA 16-bit layout:
//   lanes 0-15 : K = {0..7, 16..23}, lanes 16-31 : K = {8..15, 24..31}
__device__ __forceinline__ v16bf load_frag_rowmajor(const __bf16* base, int ld, int lane) {
  int row = lane & 15;
  int kb  = (lane & 16) ? 8 : 0;
  const __bf16* p = base + row * ld + kb;
  v8bf lo = *(const v8bf*)(p);       // K = kb .. kb+7
  v8bf hi = *(const v8bf*)(p + 16);  // K = kb+16 .. kb+23
  v16bf r;
#pragma unroll
  for (int i = 0; i < 8; ++i) { r[i] = lo[i]; r[i + 8] = hi[i]; }
  return r;
}

// B-fragment (32x16, K x N) from a K-major (strided) source: elem(k,n) = base[k*ld + n]
__device__ __forceinline__ v16bf load_frag_strided(const __bf16* base, int ld, int lane) {
  int n  = lane & 15;
  int kb = (lane & 16) ? 8 : 0;
  v16bf r;
#pragma unroll
  for (int i = 0; i < 8; ++i) {
    r[i]     = base[(kb + i) * ld + n];
    r[i + 8] = base[(kb + 16 + i) * ld + n];
  }
  return r;
}

__device__ __forceinline__ v8f wmma_bf16(v16bf a, v16bf b, v8f c) {
  return __builtin_amdgcn_wmma_f32_16x16x32_bf16(false, a, false, b, (short)0, c, false, false);
}
// same-B back-to-back: operand-cache reuse hint on B
__device__ __forceinline__ v8f wmma_bf16_rb(v16bf a, v16bf b, v8f c) {
  return __builtin_amdgcn_wmma_f32_16x16x32_bf16(false, a, false, b, (short)0, c, false, true);
}
// same-A back-to-back: operand-cache reuse hint on A
__device__ __forceinline__ v8f wmma_bf16_ra(v16bf a, v16bf b, v8f c) {
  return __builtin_amdgcn_wmma_f32_16x16x32_bf16(false, a, false, b, (short)0, c, true, false);
}

// ---------------- prep: weights -> bf16, bias -> transposed bf16 table ----------------
// bias is pre-scaled by log2(e) so softmax can use the raw hardware base-2 exp.
__global__ void winmsa_prep(const float* __restrict__ Wqkv, const float* __restrict__ Wproj,
                            const float* __restrict__ rpe, const int* __restrict__ rpe_idx,
                            __bf16* __restrict__ wq, __bf16* __restrict__ wp,
                            __bf16* __restrict__ biasT) {
  int i = blockIdx.x * blockDim.x + threadIdx.x;
  int stride = gridDim.x * blockDim.x;
  for (int t = i; t < 3 * C_ * C_; t += stride) wq[t] = (__bf16)Wqkv[t];
  for (int t = i; t < C_ * C_; t += stride)     wp[t] = (__bf16)Wproj[t];
  // biasT[h][key m][query n] = log2e * rpe[rpe_idx[n][m]][h]
  for (int t = i; t < NH_ * N_ * N_; t += stride) {
    int h   = t >> 12;
    int rem = t & 4095;
    int m   = rem >> 6;   // key
    int n   = rem & 63;   // query
    biasT[t] = (__bf16)(rpe[rpe_idx[n * 64 + m] * NH_ + h] * LOG2E);
  }
}

// ---------------- fused window MSA: one block per window, 8 waves ----------------
__global__ __launch_bounds__(256)
void winmsa_fused(const float* __restrict__ x,
                  const __bf16* __restrict__ wq,    // [768][256] bf16
                  const __bf16* __restrict__ wp,    // [256][256] bf16
                  const __bf16* __restrict__ biasT, // [8][64 key][64 query], pre-scaled
                  float* __restrict__ out) {
  __shared__ __align__(16) __bf16 smem[(XS_LD + QKV_LD) * N_];
  __bf16* xs  = smem;               // 64 x 256 (stride XS_LD); reused for merged O
  __bf16* qkv = smem + XS_LD * N_;  // 64 x 768 (stride QKV_LD): q|k|v

  const int tid  = threadIdx.x;
  const int lane = tid & 31;
  const int wv   = tid >> 5;
  const long win = blockIdx.x;
  const float* xw = x + win * (N_ * C_);

  // ---- stage x window into LDS as bf16 (vectorized: b128 loads, b64 stores) ----
  {
    const float4* xw4 = (const float4*)xw;
    for (int t = tid; t < (N_ * C_) / 4; t += 256) {
      float4 f = xw4[t];
      int row = t >> 6;
      int c4  = (t & 63) << 2;
      v4bf b;
      b.x = (__bf16)f.x; b.y = (__bf16)f.y; b.z = (__bf16)f.z; b.w = (__bf16)f.w;
      *(v4bf*)(xs + row * XS_LD + c4) = b;
    }
  }
  __syncthreads();

  const int cNc = (lane & 15);
  const int cMb = (lane & 16) ? 8 : 0;

  // ---- QKV GEMM: qkv[64][768] = xs @ Wqkv^T ----
  // B-stationary over mt: every weight fragment is loaded ONCE per block and
  // feeds 4 independent accumulators (4 back-to-back WMMAs, reuse-B hint).
  {
    v16bf A[4][8];
#pragma unroll
    for (int mt = 0; mt < 4; ++mt)
#pragma unroll
      for (int kk = 0; kk < 8; ++kk)
        A[mt][kk] = load_frag_rowmajor(xs + (mt * 16) * XS_LD + kk * 32, XS_LD, lane);

    for (int j = 0; j < 6; ++j) {
      int nt = wv * 6 + j;
      v8f acc[4];
      {
        v8f z = {};
#pragma unroll
        for (int mt = 0; mt < 4; ++mt) acc[mt] = z;
      }
#pragma unroll
      for (int kk = 0; kk < 8; ++kk) {
        v16bf b = load_frag_rowmajor(wq + (nt * 16) * C_ + kk * 32, C_, lane);
        acc[0] = wmma_bf16(A[0][kk], b, acc[0]);
        acc[1] = wmma_bf16_rb(A[1][kk], b, acc[1]);
        acc[2] = wmma_bf16_rb(A[2][kk], b, acc[2]);
        acc[3] = wmma_bf16_rb(A[3][kk], b, acc[3]);
      }
#pragma unroll
      for (int mt = 0; mt < 4; ++mt) {
#pragma unroll
        for (int r = 0; r < 8; ++r)
          qkv[(mt * 16 + cMb + r) * QKV_LD + nt * 16 + cNc] = (__bf16)acc[mt][r];
      }
    }
  }
  __syncthreads();

  // ---- attention: one head per wave ----
  const int h  = wv;
  const int qo = h * HD_;
  const int ko = C_ + h * HD_;
  const int vo = 2 * C_ + h * HD_;
  const float scale2 = 0.17677669529663688f * LOG2E;  // (1/sqrt(32)) * log2(e)

  // S^T[key][query] = K @ Q^T   (C layout: lane = query, VGPR = key)
  v8f St[4][4];  // [key tile][query tile]
  {
    v16bf qb[4];
#pragma unroll
    for (int qt = 0; qt < 4; ++qt)
      qb[qt] = load_frag_rowmajor(qkv + (qt * 16) * QKV_LD + qo, QKV_LD, lane);
#pragma unroll
    for (int kt = 0; kt < 4; ++kt) {
      v16bf ka = load_frag_rowmajor(qkv + (kt * 16) * QKV_LD + ko, QKV_LD, lane);
      v8f z = {};
      St[kt][0] = wmma_bf16(ka, qb[0], z);
      St[kt][1] = wmma_bf16_ra(ka, qb[1], z);
      St[kt][2] = wmma_bf16_ra(ka, qb[2], z);
      St[kt][3] = wmma_bf16_ra(ka, qb[3], z);
    }
  }

  // scale + relative-position bias (both pre-scaled by log2e), softmax via hw exp2
  {
    const __bf16* bh = biasT + h * (N_ * N_);
#pragma unroll
    for (int kt = 0; kt < 4; ++kt)
#pragma unroll
      for (int qt = 0; qt < 4; ++qt)
#pragma unroll
        for (int r = 0; r < 8; ++r)
          St[kt][qt][r] = St[kt][qt][r] * scale2 +
                          (float)bh[(kt * 16 + cMb + r) * 64 + qt * 16 + cNc];

#pragma unroll
    for (int qt = 0; qt < 4; ++qt) {
      float m = -3.4e38f;
#pragma unroll
      for (int kt = 0; kt < 4; ++kt)
#pragma unroll
        for (int r = 0; r < 8; ++r) m = fmaxf(m, St[kt][qt][r]);
      m = fmaxf(m, __shfl_xor(m, 16, 32));  // combine key halves (lanes l / l^16)
      float s = 0.f;
#pragma unroll
      for (int kt = 0; kt < 4; ++kt)
#pragma unroll
        for (int r = 0; r < 8; ++r) {
          float p = __builtin_amdgcn_exp2f(St[kt][qt][r] - m);  // bare v_exp_f32
          St[kt][qt][r] = p;
          s += p;
        }
      s += __shfl_xor(s, 16, 32);
      float inv = 1.f / s;
#pragma unroll
      for (int kt = 0; kt < 4; ++kt)
#pragma unroll
        for (int r = 0; r < 8; ++r) St[kt][qt][r] *= inv;
    }
  }

  // O^T[d][q] = V^T @ P^T.  B-fragment of P^T is a pure register repack of St.
  v8f Ot[2][4];
  {
    v8f z = {};
#pragma unroll
    for (int dt = 0; dt < 2; ++dt)
#pragma unroll
      for (int qt = 0; qt < 4; ++qt) Ot[dt][qt] = z;
  }
#pragma unroll
  for (int kf = 0; kf < 2; ++kf) {
    v16bf va[2];
#pragma unroll
    for (int dt = 0; dt < 2; ++dt)
      va[dt] = load_frag_strided(qkv + (kf * 32) * QKV_LD + vo + dt * 16, QKV_LD, lane);
#pragma unroll
    for (int qt = 0; qt < 4; ++qt) {
      v16bf pb;
#pragma unroll
      for (int i = 0; i < 8; ++i) {
        pb[i]     = (__bf16)St[kf * 2][qt][i];      // keys kf*32 + {0..7 | 8..15}
        pb[i + 8] = (__bf16)St[kf * 2 + 1][qt][i];  // keys kf*32 + {16..23 | 24..31}
      }
      Ot[0][qt] = wmma_bf16(va[0], pb, Ot[0][qt]);
      Ot[1][qt] = wmma_bf16_rb(va[1], pb, Ot[1][qt]);
    }
  }

  // store merged heads O[query][h*32+d] into the (now dead) xs region
  {
#pragma unroll
    for (int dt = 0; dt < 2; ++dt)
#pragma unroll
      for (int qt = 0; qt < 4; ++qt)
#pragma unroll
        for (int r = 0; r < 8; ++r)
          xs[(qt * 16 + cNc) * XS_LD + h * HD_ + dt * 16 + cMb + r] = (__bf16)Ot[dt][qt][r];
  }
  __syncthreads();

  // ---- proj GEMM: out[64][256] = O @ Wproj^T (B-stationary over mt) ----
  float* ow = out + win * (N_ * C_);
  {
    v16bf A[4][8];
#pragma unroll
    for (int mt = 0; mt < 4; ++mt)
#pragma unroll
      for (int kk = 0; kk < 8; ++kk)
        A[mt][kk] = load_frag_rowmajor(xs + (mt * 16) * XS_LD + kk * 32, XS_LD, lane);

#pragma unroll
    for (int j = 0; j < 2; ++j) {
      int nt = wv * 2 + j;
      v8f acc[4];
      {
        v8f z = {};
#pragma unroll
        for (int mt = 0; mt < 4; ++mt) acc[mt] = z;
      }
#pragma unroll
      for (int kk = 0; kk < 8; ++kk) {
        v16bf b = load_frag_rowmajor(wp + (nt * 16) * C_ + kk * 32, C_, lane);
        acc[0] = wmma_bf16(A[0][kk], b, acc[0]);
        acc[1] = wmma_bf16_rb(A[1][kk], b, acc[1]);
        acc[2] = wmma_bf16_rb(A[2][kk], b, acc[2]);
        acc[3] = wmma_bf16_rb(A[3][kk], b, acc[3]);
      }
#pragma unroll
      for (int mt = 0; mt < 4; ++mt) {
#pragma unroll
        for (int r = 0; r < 8; ++r)
          ow[(mt * 16 + cMb + r) * C_ + nt * 16 + cNc] = acc[mt][r];
      }
    }
  }
}

extern "C" void kernel_launch(void* const* d_in, const int* in_sizes, int n_in,
                              void* d_out, int out_size, void* d_ws, size_t ws_size,
                              hipStream_t stream) {
  const float* x     = (const float*)d_in[0];
  const float* Wqkv  = (const float*)d_in[1];
  const float* Wproj = (const float*)d_in[2];
  const float* rpe   = (const float*)d_in[3];
  const int*   ridx  = (const int*)d_in[4];
  float* out = (float*)d_out;

  char* ws = (char*)d_ws;
  __bf16* wq    = (__bf16*)ws;                         // 768*256*2  = 393216 B
  __bf16* wp    = (__bf16*)(ws + 393216);              // 256*256*2  = 131072 B
  __bf16* biasT = (__bf16*)(ws + 393216 + 131072);     // 8*64*64*2  = 65536 B

  winmsa_prep<<<256, 256, 0, stream>>>(Wqkv, Wproj, rpe, ridx, wq, wp, biasT);
  winmsa_fused<<<4096, 256, 0, stream>>>(x, wq, wp, biasT, out);
}
